// objCrossAttention_51505247814065
// MI455X (gfx1250) — matrix operand order repaired and checked by reading, not compiled, not run on hardware
//
#include <hip/hip_runtime.h>

// Problem constants (match the reference)
#define B_      16
#define N_      4096
#define M_      256
#define QDIM    512
#define CTXDIM  768
#define HEADS   8
#define DHEAD   64
#define INNER   512   // HEADS*DHEAD

typedef __attribute__((ext_vector_type(16))) __bf16 v16bf;
typedef __attribute__((ext_vector_type(8)))  __bf16 v8bf;
typedef __attribute__((ext_vector_type(4)))  __bf16 v4bf;
typedef __attribute__((ext_vector_type(8)))  float  v8f;

// ---- fragment load helpers -------------------------------------------------
static __device__ __forceinline__ v16bf ld16(const __bf16* p) {
  v8bf lo = *(const v8bf*)p;
  v8bf hi = *(const v8bf*)(p + 8);
  v16bf r;
#pragma unroll
  for (int i = 0; i < 8; ++i) { r[i] = lo[i]; r[i + 8] = hi[i]; }
  return r;
}
static __device__ __forceinline__ v16bf ld8x2(const __bf16* p0, const __bf16* p1) {
  v8bf lo = *(const v8bf*)p0;
  v8bf hi = *(const v8bf*)p1;
  v16bf r;
#pragma unroll
  for (int i = 0; i < 8; ++i) { r[i] = lo[i]; r[i + 8] = hi[i]; }
  return r;
}
static __device__ __forceinline__ v8f wmma_bf16(v16bf a, v16bf b, v8f c) {
  return __builtin_amdgcn_wmma_f32_16x16x32_bf16(false, a, false, b, (short)0, c, false, false);
}

// ---- CDNA5 async copy: global -> LDS, 16B per lane, tracked by ASYNCcnt ----
static __device__ __forceinline__ void async_cp16(unsigned ldsOff, unsigned gByteOff,
                                                  const void* base) {
  asm volatile("global_load_async_to_lds_b128 %0, %1, %2"
               :: "v"(ldsOff), "v"(gByteOff), "s"(base) : "memory");
}
static __device__ __forceinline__ void wait_async0() {
  asm volatile("s_wait_asynccnt 0x0" ::: "memory");
}

// ---------------------------------------------------------------------------
// Elementwise f32 -> bf16 (n multiple of 8).
// ---------------------------------------------------------------------------
__global__ __launch_bounds__(256)
void cvt_bf16_kernel(const float* __restrict__ s, __bf16* __restrict__ d, int n) {
  int i = (blockIdx.x * 256 + threadIdx.x) * 8;
  if (i >= n) return;
  float4 a = *(const float4*)(s + i);
  float4 b = *(const float4*)(s + i + 4);
  v8bf o;
  o[0] = (__bf16)a.x; o[1] = (__bf16)a.y; o[2] = (__bf16)a.z; o[3] = (__bf16)a.w;
  o[4] = (__bf16)b.x; o[5] = (__bf16)b.y; o[6] = (__bf16)b.z; o[7] = (__bf16)b.w;
  *(v8bf*)(d + i) = o;
}

// ---------------------------------------------------------------------------
// Weight transpose + convert: Wt[n][k] = (bf16)W[k][n].  (small matrices)
// ---------------------------------------------------------------------------
__global__ __launch_bounds__(256)
void wtrans_bf16_kernel(const float* __restrict__ W, __bf16* __restrict__ Wt,
                        int K, int N) {
  int idx = blockIdx.x * 256 + threadIdx.x;
  if (idx >= K * N) return;
  int n = idx / K, k = idx - n * K;
  Wt[idx] = (__bf16)W[(size_t)k * N + n];
}

// ---------------------------------------------------------------------------
// WMMA GEMM, all-bf16 operands: C[M,N] = A[M,K] * Wt[N,K]^T
// 128x128 C tile per block (256 thr, 8 waves: 4x2 grid of 32x64 wave tiles).
// K-step 32. Double-buffered LDS staged with global_load_async_to_lds_b128;
// copies of step k+1 overlap the 8 WMMAs of step k. Rows padded to 40 bf16
// (80B) so b128 fragment reads are bank-conflict-free.
// ---------------------------------------------------------------------------
template <int OUT_IS_BF16, int ADD_BIAS>
__global__ __launch_bounds__(256)
void gemm_wmma_async(const __bf16* __restrict__ A, const __bf16* __restrict__ Wt,
                     void* __restrict__ Cv, const float* __restrict__ bias,
                     float scale, int M, int K, int N) {
  __shared__ __align__(16) __bf16 As[2][128 * 40];
  __shared__ __align__(16) __bf16 Ws[2][128 * 40];

  const int tid = threadIdx.x, wave = tid >> 5, lane = tid & 31;
  const int lr = lane & 15, hi = lane >> 4;
  const int m0 = blockIdx.x * 128, n0 = blockIdx.y * 128;
  const int wrow = (wave >> 1) * 32, wcol = (wave & 1) * 64;

  v8f acc[2][4] = {};

  const int KT = K >> 5;  // K-steps of 32

  // Issue async staging for k-step kt into buffer buf: 128x32 bf16 tiles
  // = 512 x 16B chunks each for A and W; 2 chunks of each per thread.
  auto issue = [&](int buf, int k0) {
#pragma unroll
    for (int i = 0; i < 2; ++i) {
      const int ch = i * 256 + tid;         // 0..511
      const int r = ch >> 2, c = (ch & 3) * 8;  // row, elem offset (chunk*8)
      async_cp16((unsigned)(uintptr_t)&As[buf][r * 40 + c],
                 (unsigned)(((m0 + r) * K + k0 + c) * 2), A);
      async_cp16((unsigned)(uintptr_t)&Ws[buf][r * 40 + c],
                 (unsigned)(((n0 + r) * K + k0 + c) * 2), Wt);
    }
  };

  issue(0, 0);
  for (int kt = 0; kt < KT; ++kt) {
    wait_async0();        // this wave's copies (incl. current buffer) complete
    __syncthreads();      // => everyone's copies complete; prior reads done
    if (kt + 1 < KT) issue((kt + 1) & 1, (kt + 1) * 32);  // overlap next stage

    const int buf = kt & 1;
    v16bf a[2];
#pragma unroll
    for (int mt = 0; mt < 2; ++mt) {
      const __bf16* ap = &As[buf][(wrow + mt * 16 + lr) * 40];
      a[mt] = ld16(ap + 16 * hi);
    }
#pragma unroll
    for (int nt = 0; nt < 4; ++nt) {
      v16bf bfrag = ld16(&Ws[buf][(wcol + nt * 16 + lr) * 40 + 16 * hi]);
#pragma unroll
      for (int mt = 0; mt < 2; ++mt)
        acc[mt][nt] = wmma_bf16(a[mt], bfrag, acc[mt][nt]);
    }
  }

  // Epilogue: C layout (M = r + 8*hi, N = nt*16 + lr).
#pragma unroll
  for (int mt = 0; mt < 2; ++mt) {
#pragma unroll
    for (int nt = 0; nt < 4; ++nt) {
#pragma unroll
      for (int r = 0; r < 8; ++r) {
        int row = m0 + wrow + mt * 16 + 8 * hi + r;
        int col = n0 + wcol + nt * 16 + lr;
        float v = acc[mt][nt][r] * scale;
        if (ADD_BIAS) v += bias[col];
        if (OUT_IS_BF16) ((__bf16*)Cv)[(size_t)row * N + col] = (__bf16)v;
        else             ((float*)Cv)[(size_t)row * N + col] = v;
      }
    }
  }
}

// ---------------------------------------------------------------------------
// Attention: per block = one (batch, head), 128 q rows (8 waves x 16).
// S = Q*K^T (scale pre-folded into Q), masked (!obj_vector -> -max, !obj_mask -> 0),
// O = S*V. V head-slice staged transposed in LDS; S tiles round-trip through
// per-wave padded LDS to convert C-layout -> A-layout.
// ---------------------------------------------------------------------------
__global__ __launch_bounds__(256)
void attn_wmma_kernel(const __bf16* __restrict__ Q, const __bf16* __restrict__ Kh,
                      const __bf16* __restrict__ Vh, const int* __restrict__ obj_mask,
                      const int* __restrict__ obj_vector, __bf16* __restrict__ O) {
  __shared__ __align__(16) __bf16 Vt[DHEAD * M_];   // 64x256 transposed V (32 KB)
  __shared__ __align__(16) __bf16 Ssh[8 * 16 * 40]; // per-wave 16x32 S tile

  const int tid = threadIdx.x, wave = tid >> 5, lane = tid & 31;
  const int lr = lane & 15, hi = lane >> 4;
  const int bh = blockIdx.y, b = bh >> 3, h = bh & 7;

  // Stage V transposed: thread t owns key-row t (one 128B line per row).
  {
    const __bf16* vrow = Vh + ((size_t)(b * M_ + tid)) * INNER + h * DHEAD;
#pragma unroll
    for (int i = 0; i < 16; ++i) {
      v4bf v = *(const v4bf*)(vrow + i * 4);
#pragma unroll
      for (int j = 0; j < 4; ++j) Vt[(i * 4 + j) * M_ + tid] = v[j];
    }
  }
  __syncthreads();

  const int qbase = blockIdx.x * 128 + wave * 16;
  const __bf16* qp = Q + ((size_t)(b * N_ + qbase + lr)) * INNER + h * DHEAD;
  v16bf a0 = ld16(qp + 16 * hi);
  v16bf a1 = ld16(qp + 32 + 16 * hi);

  v8f oacc[4] = {};
  __bf16* Sw = &Ssh[wave * 16 * 40];

  for (int mc = 0; mc < M_; mc += 32) {
#pragma unroll
    for (int t = 0; t < 2; ++t) {
      const int mb = mc + t * 16;
      const __bf16* kp = Kh + ((size_t)(b * M_ + mb + lr)) * INNER + h * DHEAD;
      v8f s = {};
      s = wmma_bf16(a0, ld16(kp + 16 * hi), s);
      s = wmma_bf16(a1, ld16(kp + 32 + 16 * hi), s);

      const int mcol = mb + lr;
      const int ov = obj_vector[b * M_ + mcol];
#pragma unroll
      for (int r = 0; r < 8; ++r) {
        float sv = s[r];
        if (!ov) sv = -3.3895314e38f;
        const int n = qbase + 8 * hi + r;
        if (!obj_mask[((size_t)(b * N_ + n)) * M_ + mcol]) sv = 0.0f;
        Sw[(8 * hi + r) * 40 + t * 16 + lr] = (__bf16)sv;
      }
    }
    const __bf16* sp = &Sw[lr * 40];
    v16bf sa = ld16(sp + 16 * hi);
#pragma unroll
    for (int nt = 0; nt < 4; ++nt) {
      v16bf vb = ld16(&Vt[(nt * 16 + lr) * M_ + mc + 16 * hi]);
      oacc[nt] = wmma_bf16(sa, vb, oacc[nt]);
    }
  }

  __bf16* op = O + ((size_t)(b * N_ + qbase)) * INNER + h * DHEAD;
#pragma unroll
  for (int nt = 0; nt < 4; ++nt)
#pragma unroll
    for (int r = 0; r < 8; ++r)
      op[(size_t)(8 * hi + r) * INNER + nt * 16 + lr] = (__bf16)oacc[nt][r];
}

// ---------------------------------------------------------------------------
extern "C" void kernel_launch(void* const* d_in, const int* in_sizes, int n_in,
                              void* d_out, int out_size, void* d_ws, size_t ws_size,
                              hipStream_t stream) {
  (void)in_sizes; (void)n_in; (void)out_size; (void)ws_size;
  const float* x          = (const float*)d_in[0];
  const float* obj_txt    = (const float*)d_in[1];
  const int*   obj_mask   = (const int*)d_in[2];
  const int*   obj_vector = (const int*)d_in[3];
  const float* Wq         = (const float*)d_in[4];
  const float* Wk         = (const float*)d_in[5];
  const float* Wv         = (const float*)d_in[6];
  const float* Wo         = (const float*)d_in[7];
  const float* bo         = (const float*)d_in[8];

  // Workspace layout (all bf16 elements).
  __bf16* Qb  = (__bf16*)d_ws;
  __bf16* Kb  = Qb  + (size_t)B_ * N_ * INNER;
  __bf16* Vb  = Kb  + (size_t)B_ * M_ * INNER;
  __bf16* Ob  = Vb  + (size_t)B_ * M_ * INNER;
  __bf16* xb  = Ob  + (size_t)B_ * N_ * INNER;     // x as bf16
  __bf16* ctb = xb  + (size_t)B_ * N_ * QDIM;      // obj_txt as bf16
  __bf16* Wqt = ctb + (size_t)B_ * M_ * CTXDIM;    // [INNER][QDIM]
  __bf16* Wkt = Wqt + (size_t)INNER * QDIM;        // [INNER][CTXDIM]
  __bf16* Wvt = Wkt + (size_t)INNER * CTXDIM;      // [INNER][CTXDIM]
  __bf16* Wot = Wvt + (size_t)INNER * CTXDIM;      // [QDIM][INNER]

  const float SCALE = 0.125f;  // DIM_HEAD^-0.5, folded into Q epilogue

  // One-shot converts / weight transposes (bf16, [N][K]).
  {
    int nx = B_ * N_ * QDIM, nc = B_ * M_ * CTXDIM;
    cvt_bf16_kernel<<<nx / (256 * 8), 256, 0, stream>>>(x, xb, nx);
    cvt_bf16_kernel<<<nc / (256 * 8), 256, 0, stream>>>(obj_txt, ctb, nc);
    wtrans_bf16_kernel<<<(QDIM * INNER + 255) / 256, 256, 0, stream>>>(Wq, Wqt, QDIM, INNER);
    wtrans_bf16_kernel<<<(CTXDIM * INNER + 255) / 256, 256, 0, stream>>>(Wk, Wkt, CTXDIM, INNER);
    wtrans_bf16_kernel<<<(CTXDIM * INNER + 255) / 256, 256, 0, stream>>>(Wv, Wvt, CTXDIM, INNER);
    wtrans_bf16_kernel<<<(INNER * QDIM + 255) / 256, 256, 0, stream>>>(Wo, Wot, INNER, QDIM);
  }

  dim3 blk(256);
  // Projections (async-staged bf16 WMMA GEMMs).
  gemm_wmma_async<1, 0><<<dim3((B_ * N_) / 128, INNER / 128), blk, 0, stream>>>(
      xb, Wqt, Qb, nullptr, SCALE, B_ * N_, QDIM, INNER);
  gemm_wmma_async<1, 0><<<dim3((B_ * M_) / 128, INNER / 128), blk, 0, stream>>>(
      ctb, Wkt, Kb, nullptr, 1.0f, B_ * M_, CTXDIM, INNER);
  gemm_wmma_async<1, 0><<<dim3((B_ * M_) / 128, INNER / 128), blk, 0, stream>>>(
      ctb, Wvt, Vb, nullptr, 1.0f, B_ * M_, CTXDIM, INNER);

  // Attention (masked, no softmax per reference semantics).
  attn_wmma_kernel<<<dim3(N_ / 128, B_ * HEADS), 256, 0, stream>>>(
      Qb, Kb, Vb, obj_mask, obj_vector, Ob);

  // Output projection (bf16 A, f32 out, + bias).
  gemm_wmma_async<0, 1><<<dim3((B_ * N_) / 128, QDIM / 128), blk, 0, stream>>>(
      Ob, Wot, d_out, bo, 1.0f, B_ * N_, INNER, QDIM);
}
